// MultiHeadSelfAttention_31026843746703
// MI455X (gfx1250) — matrix operand order repaired and checked by reading, not compiled, not run on hardware
//
#include <hip/hip_runtime.h>

#define DMODEL 512
#define NHEADS 8
#define DHEAD  64
#define BATCH  4
#define SEQ    2048
#define MTOT   (BATCH * SEQ)           // 8192 rows
#define ATT_SCALE 0.04419417382415922f // 1/sqrt(512)

typedef __attribute__((ext_vector_type(16))) __bf16 v16bf;
typedef __attribute__((ext_vector_type(8)))  float  v8f;

struct Frag { int4 lo, hi; };          // 32 bytes = 16 bf16

static __device__ __forceinline__ v16bf as_v16bf(Frag f) {
  return __builtin_bit_cast(v16bf, f);
}

static __device__ __forceinline__ unsigned short f2bf(float f) {
  unsigned u = __float_as_uint(f);
  u += 0x7FFFu + ((u >> 16) & 1u);     // round-to-nearest-even
  return (unsigned short)(u >> 16);
}

static __device__ __forceinline__ v8f zero8() {
  v8f z = {0.f, 0.f, 0.f, 0.f, 0.f, 0.f, 0.f, 0.f};
  return z;
}

// A-fragment (16 x 32 bf16). Per ISA 7.12.2: lanes 0-15 -> row L, K = ks+0..7
// (regs 0-3) and ks+16..23 (regs 4-7); lanes 16-31 -> row L-16, K = ks+8..15 /
// ks+24..31. Source is row-major along K with leading dim ld -> two int4 loads.
static __device__ __forceinline__ Frag load_a(const unsigned short* base, int row0,
                                              int ld, int ks, int lane) {
  int r = row0 + (lane & 15);
  int k = ks + ((lane >> 4) << 3);
  const int4* p = (const int4*)(base + (size_t)r * ld + k);
  Frag f; f.lo = p[0]; f.hi = p[2];    // +16 elements
  return f;
}

// B-fragment (32 x 16 bf16). Same K order as A: lanes 0-15 -> col L, K = ks+0..7
// and ks+16..23; lanes 16-31 -> col L-16, K = ks+8..15 / ks+24..31. Source stores
// each output-column's K-run contiguously (leading dim ld) -> two int4 loads.
static __device__ __forceinline__ Frag load_b(const unsigned short* base, int col0,
                                              int ld, int ks, int lane) {
  int c = col0 + (lane & 15);
  int k = ks + ((lane >> 4) << 3);
  const int4* p = (const int4*)(base + (size_t)c * ld + k);
  Frag f; f.lo = p[0]; f.hi = p[2];    // +16 elements
  return f;
}

static __device__ __forceinline__ v8f wmma_bf(Frag a, Frag b, v8f c) {
  return __builtin_amdgcn_wmma_f32_16x16x32_bf16(
      false, as_v16bf(a), false, as_v16bf(b), (short)0, c, false, false);
}

// ---------------------------------------------------------------- converts ---
__global__ void k_cvt_bf16(const float* __restrict__ src,
                           unsigned short* __restrict__ dst, int n) {
  int i = blockIdx.x * blockDim.x + threadIdx.x;
  int stride = gridDim.x * blockDim.x;
  for (; i < n; i += stride) dst[i] = f2bf(src[i]);
}

// W [K=512][N=512] f32 -> Wt [N][K] bf16 (so B-fragments load contiguously)
__global__ void k_cvt_wT(const float* __restrict__ w,
                         unsigned short* __restrict__ wt) {
  int i = blockIdx.x * blockDim.x + threadIdx.x;  // 0..512*512-1
  int n = i & (DMODEL - 1), k = i >> 9;
  wt[(size_t)n * DMODEL + k] = f2bf(w[(size_t)k * DMODEL + n]);
}

// ------------------------------------------------------------------- GEMM ----
// C[M=8192, N=512] = A[M,512] @ W (W given transposed as Bt[N,K]), + bias.
// MODE 0: bf16 row-major out. MODE 1: bf16 per-head-transposed out
// ([B,H,Dh,S], for V). MODE 2: f32 row-major out (final projection).
// Wave tile 32x64: 2 A-frags x 4 B-frags -> 8 WMMAs per K=32 step, 16 steps.
template <int MODE>
__global__ __launch_bounds__(128) void k_gemm(
    const unsigned short* __restrict__ A, const unsigned short* __restrict__ Bt,
    const float* __restrict__ bias, unsigned short* __restrict__ Cb,
    float* __restrict__ Cf) {
  const int lane = threadIdx.x & 31;
  const int wid  = blockIdx.x * (blockDim.x >> 5) + (threadIdx.x >> 5);
  const int ntiles = DMODEL / 64;                  // 8
  const int m0 = (wid / ntiles) * 32;
  const int n0 = (wid % ntiles) * 64;

  v8f acc[2][4];
  for (int i = 0; i < 2; i++)
    for (int t = 0; t < 4; t++) acc[i][t] = zero8();

  for (int k = 0; k < DMODEL; k += 32) {
    Frag a0 = load_a(A, m0,      DMODEL, k, lane);
    Frag a1 = load_a(A, m0 + 16, DMODEL, k, lane);
    Frag b0 = load_b(Bt, n0,      DMODEL, k, lane);
    Frag b1 = load_b(Bt, n0 + 16, DMODEL, k, lane);
    Frag b2 = load_b(Bt, n0 + 32, DMODEL, k, lane);
    Frag b3 = load_b(Bt, n0 + 48, DMODEL, k, lane);
    acc[0][0] = wmma_bf(a0, b0, acc[0][0]);
    acc[0][1] = wmma_bf(a0, b1, acc[0][1]);
    acc[0][2] = wmma_bf(a0, b2, acc[0][2]);
    acc[0][3] = wmma_bf(a0, b3, acc[0][3]);
    acc[1][0] = wmma_bf(a1, b0, acc[1][0]);
    acc[1][1] = wmma_bf(a1, b1, acc[1][1]);
    acc[1][2] = wmma_bf(a1, b2, acc[1][2]);
    acc[1][3] = wmma_bf(a1, b3, acc[1][3]);
  }

  const int half = lane >> 4;                      // C-layout: reg r = row r (+8)
  const int lc   = lane & 15;                      // C-layout: lane = col
  for (int mi = 0; mi < 2; mi++) {
    const int mrow0 = m0 + mi * 16 + half * 8;     // 8 consecutive rows in regs
    for (int t = 0; t < 4; t++) {
      const int n = n0 + t * 16 + lc;
      const float bv = bias[n];
      if (MODE == 0) {
        for (int r = 0; r < 8; r++)
          Cb[(size_t)(mrow0 + r) * DMODEL + n] = f2bf(acc[mi][t][r] + bv);
      } else if (MODE == 2) {
        for (int r = 0; r < 8; r++)
          Cf[(size_t)(mrow0 + r) * DMODEL + n] = acc[mi][t][r] + bv;
      } else {  // MODE 1: Vt[((b*H+h)*Dh + d) * SEQ + s], 8 rows pack to int4
        const int h = n >> 6, d = n & (DHEAD - 1);
        const int b = mrow0 >> 11, s = mrow0 & (SEQ - 1);
        unsigned short pk[8];
        for (int r = 0; r < 8; r++) pk[r] = f2bf(acc[mi][t][r] + bv);
        int4 v;
        v.x = (int)pk[0] | ((int)pk[1] << 16);
        v.y = (int)pk[2] | ((int)pk[3] << 16);
        v.z = (int)pk[4] | ((int)pk[5] << 16);
        v.w = (int)pk[6] | ((int)pk[7] << 16);
        *(int4*)(Cb + ((size_t)((b * NHEADS + h) * DHEAD + d) * SEQ + s)) = v;
      }
    }
  }
}

// -------------------------------------------------------- flash attention ----
// One wave = 16 query rows of one (b,h). Streams keys in blocks of 32 with an
// online softmax; P is moved D-layout -> A-layout through a 1KB LDS tile.
__global__ __launch_bounds__(128) void k_attention(
    const unsigned short* __restrict__ qb,   // [B*S, 512] bf16
    const unsigned short* __restrict__ kb,   // [B*S, 512] bf16
    const unsigned short* __restrict__ vt,   // [B*H*Dh, S] bf16 (transposed V)
    unsigned short* __restrict__ att) {      // [B*S, 512] bf16
  __shared__ __align__(16) unsigned short pl[4][16][32];
  const int lane  = threadIdx.x & 31;
  const int wslot = threadIdx.x >> 5;
  const int wid   = blockIdx.x * 4 + wslot;        // 0..4095
  const int qtiles = SEQ / 16;                     // 128
  const int qt = wid % qtiles;
  const int bh = wid / qtiles;                     // 0..31
  const int h = bh % NHEADS, b = bh / NHEADS;
  const int s0 = qt * 16;

  const unsigned short* qptr = qb + (size_t)(b * SEQ) * DMODEL + h * DHEAD;
  const unsigned short* kptr = kb + (size_t)(b * SEQ) * DMODEL + h * DHEAD;
  const unsigned short* vptr = vt + (size_t)(b * NHEADS + h) * DHEAD * SEQ;

  Frag qa0 = load_a(qptr, s0, DMODEL, 0,  lane);   // K(dhead) 0..31
  Frag qa1 = load_a(qptr, s0, DMODEL, 32, lane);   // K(dhead) 32..63

  v8f acc[4];
  for (int t = 0; t < 4; t++) acc[t] = zero8();
  float m[8], l[8];
  for (int r = 0; r < 8; r++) { m[r] = -3.0e38f; l[r] = 0.0f; }

  const int half = lane >> 4;
  const int lc   = lane & 15;

  for (int j0 = 0; j0 < SEQ; j0 += 32) {
    // scores: two 16x16 tiles (keys j0..15, j0+16..31), contraction K=64
    v8f sc0 = zero8(), sc1 = zero8();
    Frag kf00 = load_b(kptr, j0,      DMODEL, 0,  lane);
    Frag kf01 = load_b(kptr, j0,      DMODEL, 32, lane);
    Frag kf10 = load_b(kptr, j0 + 16, DMODEL, 0,  lane);
    Frag kf11 = load_b(kptr, j0 + 16, DMODEL, 32, lane);
    sc0 = wmma_bf(qa0, kf00, sc0);
    sc0 = wmma_bf(qa1, kf01, sc0);
    sc1 = wmma_bf(qa0, kf10, sc1);
    sc1 = wmma_bf(qa1, kf11, sc1);

    // online softmax; row of reg r spans the 16 lanes of this half
    float p0[8], p1[8];
    for (int r = 0; r < 8; r++) {
      float a0 = sc0[r] * ATT_SCALE;
      float a1 = sc1[r] * ATT_SCALE;
      float mx = fmaxf(a0, a1);
      mx = fmaxf(mx, __shfl_xor(mx, 1));
      mx = fmaxf(mx, __shfl_xor(mx, 2));
      mx = fmaxf(mx, __shfl_xor(mx, 4));
      mx = fmaxf(mx, __shfl_xor(mx, 8));
      float mnew = fmaxf(m[r], mx);
      float corr = __expf(m[r] - mnew);
      m[r] = mnew;
      p0[r] = __expf(a0 - mnew);
      p1[r] = __expf(a1 - mnew);
      float srow = p0[r] + p1[r];
      srow += __shfl_xor(srow, 1);
      srow += __shfl_xor(srow, 2);
      srow += __shfl_xor(srow, 4);
      srow += __shfl_xor(srow, 8);
      l[r] = l[r] * corr + srow;
      for (int t = 0; t < 4; t++) acc[t][r] *= corr;
    }

    // P: D-layout regs -> LDS (row,key) -> A-fragment (in-wave, dscnt wait)
    for (int r = 0; r < 8; r++) {
      int row = r + half * 8;
      pl[wslot][row][lc]      = f2bf(p0[r]);
      pl[wslot][row][lc + 16] = f2bf(p1[r]);
    }
    asm volatile("s_wait_dscnt 0x0" ::: "memory");
    Frag pa;
    {
      int r = lane & 15;
      int k = (lane >> 4) << 3;
      const int4* p = (const int4*)(&pl[wslot][r][k]);
      pa.lo = p[0];
      pa.hi = p[2];
    }

    // acc += P @ V   (V transposed: per-lane contiguous along key index)
    Frag vb0 = load_b(vptr, 0,  SEQ, j0, lane);
    Frag vb1 = load_b(vptr, 16, SEQ, j0, lane);
    Frag vb2 = load_b(vptr, 32, SEQ, j0, lane);
    Frag vb3 = load_b(vptr, 48, SEQ, j0, lane);
    acc[0] = wmma_bf(pa, vb0, acc[0]);
    acc[1] = wmma_bf(pa, vb1, acc[1]);
    acc[2] = wmma_bf(pa, vb2, acc[2]);
    acc[3] = wmma_bf(pa, vb3, acc[3]);
  }

  // normalize and store att (bf16 row-major, input to the Wo GEMM)
  unsigned short* optr =
      att + (size_t)(b * SEQ + s0 + half * 8) * DMODEL + h * DHEAD;
  float inv[8];
  for (int r = 0; r < 8; r++) inv[r] = 1.0f / l[r];
  for (int t = 0; t < 4; t++)
    for (int r = 0; r < 8; r++)
      optr[(size_t)r * DMODEL + t * 16 + lc] = f2bf(acc[t][r] * inv[r]);
}

// ---------------------------------------------------------------- launch -----
extern "C" void kernel_launch(void* const* d_in, const int* in_sizes, int n_in,
                              void* d_out, int out_size, void* d_ws,
                              size_t ws_size, hipStream_t stream) {
  const float* x  = (const float*)d_in[0];
  const float* Wq = (const float*)d_in[1];
  const float* bq = (const float*)d_in[2];
  const float* Wk = (const float*)d_in[3];
  const float* bk = (const float*)d_in[4];
  const float* Wv = (const float*)d_in[5];
  const float* bv = (const float*)d_in[6];
  const float* Wo = (const float*)d_in[7];
  const float* bo = (const float*)d_in[8];
  float* out = (float*)d_out;

  char* ws = (char*)d_ws;
  size_t off = 0;
  auto carve = [&](size_t bytes) {
    void* p = ws + off;
    off = (off + bytes + 255) & ~(size_t)255;
    return p;
  };
  const size_t actN = (size_t)MTOT * DMODEL;       // 8192*512
  unsigned short* xb  = (unsigned short*)carve(actN * 2);
  unsigned short* wqt = (unsigned short*)carve((size_t)DMODEL * DMODEL * 2);
  unsigned short* wkt = (unsigned short*)carve((size_t)DMODEL * DMODEL * 2);
  unsigned short* wvt = (unsigned short*)carve((size_t)DMODEL * DMODEL * 2);
  unsigned short* wot = (unsigned short*)carve((size_t)DMODEL * DMODEL * 2);
  unsigned short* q   = (unsigned short*)carve(actN * 2);
  unsigned short* k   = (unsigned short*)carve(actN * 2);
  unsigned short* vT  = (unsigned short*)carve(actN * 2);   // [B,H,Dh,S]
  unsigned short* att = (unsigned short*)carve(actN * 2);

  // stage bf16 operands
  k_cvt_bf16<<<2048, 256, 0, stream>>>(x, xb, (int)actN);
  const int wThreads = DMODEL * DMODEL, wBlocks = wThreads / 256;
  k_cvt_wT<<<wBlocks, 256, 0, stream>>>(Wq, wqt);
  k_cvt_wT<<<wBlocks, 256, 0, stream>>>(Wk, wkt);
  k_cvt_wT<<<wBlocks, 256, 0, stream>>>(Wv, wvt);
  k_cvt_wT<<<wBlocks, 256, 0, stream>>>(Wo, wot);

  // projections: (M/32)*(N/64) = 2048 waves, 4 waves/block
  const int gemmBlocks = (MTOT / 32) * (DMODEL / 64) / 4;   // 512
  k_gemm<0><<<gemmBlocks, 128, 0, stream>>>(xb, wqt, bq, q, nullptr);
  k_gemm<0><<<gemmBlocks, 128, 0, stream>>>(xb, wkt, bk, k, nullptr);
  k_gemm<1><<<gemmBlocks, 128, 0, stream>>>(xb, wvt, bv, vT, nullptr);

  // attention: B*H*(S/16) = 4096 waves, 4 waves/block
  k_attention<<<1024, 128, 0, stream>>>(q, k, vT, att);

  // output projection (f32 out + bias)
  k_gemm<2><<<gemmBlocks, 128, 0, stream>>>(att, wot, bo, nullptr, out);
}
